// UltraLocationPredictor_53558242181863
// MI455X (gfx1250) — matrix operand order repaired and checked by reading, not compiled, not run on hardware
//
#include <hip/hip_runtime.h>
#include <math.h>

// ---------------------------------------------------------------------------
// Model constants (UltraLocationPredictor)
// ---------------------------------------------------------------------------
#define BB    2048
#define LL    60
#define LP    64            // L padded to 4 x 16 WMMA row tiles
#define DD    80
#define HH    8
#define HDIM  10
#define NLAY  3
#define NLOC  1187
#define DFF   160
#define INDIM 138
#define AS    100           // padded LDS row stride (floats) for K=80 A panels

// ---------------------------------------------------------------------------
// CDNA5 WMMA types (wave32, v_wmma_f32_16x16x32_bf16)
// A: 16x32 bf16 per wave, B: 32x16 bf16 per wave, C/D: 16x16 f32 (8 VGPRs)
// ---------------------------------------------------------------------------
typedef __attribute__((ext_vector_type(16))) __bf16 v16bf;
typedef __attribute__((ext_vector_type(8)))  float  v8f;

#define WMMA_BF16(a, b, c) \
  __builtin_amdgcn_wmma_f32_16x16x32_bf16(false, (a), false, (b), (short)0, (c), false, false)

// A-matrix 16x32 (MxK), 16-bit layout, NO guards (caller pads source):
//   lanes 0-15: M=lane,  elems 0-7 -> K=0..7,  elems 8-15 -> K=16..23
//   lanes16-31: M=lane-16, elems 0-7 -> K=8..15, elems 8-15 -> K=24..31
__device__ __forceinline__ v16bf load_a(const float* __restrict__ src, int ld,
                                        int lane) {
  const int m  = lane & 15;
  const int kb = (lane & 16) ? 8 : 0;
  const float* p = src + (size_t)m * ld + kb;
  v16bf a;
#pragma unroll
  for (int i = 0; i < 8; ++i)  a[i] = (__bf16)p[i];        // K = kb+0..7
#pragma unroll
  for (int i = 8; i < 16; ++i) a[i] = (__bf16)p[i + 8];    // K = kb+16..23
  return a;
}

// Packed-B GEMM: C[64][N](LDS f32) (+)= A[64][K](LDS f32, padded) x Bpacked.
// Bpacked layout: v16bf tile-step ts = nt*ksteps+ks, vector index ts*32+lane
// (one aligned 32-byte load per lane -> 2x global_load_b128, no guards).
__device__ __forceinline__ void gemm64p(const float* __restrict__ A, int lda,
                                        const v16bf* __restrict__ Bp,
                                        int ntiles, int ksteps,
                                        float* __restrict__ C, int ldc, bool add,
                                        int wave, int lane) {
  for (int t = wave; t < 4 * ntiles; t += 8) {
    const int mt = t / ntiles;
    const int nt = t - mt * ntiles;
    const v16bf* bp = Bp + ((size_t)nt * ksteps) * 32 + lane;
    const float* ap = A + (size_t)mt * 16 * lda;
    v8f acc = {};
    for (int ks = 0; ks < ksteps; ++ks) {
      v16bf a = load_a(ap + ks * 32, lda, lane);
      v16bf b = bp[(size_t)ks * 32];
      acc = WMMA_BF16(a, b, acc);
    }
    const int n  = lane & 15;
    const int mb = mt * 16 + ((lane & 16) ? 8 : 0);
    float* cp = &C[(size_t)mb * ldc + nt * 16 + n];
#pragma unroll
    for (int r = 0; r < 8; ++r) {
      float* p = cp + (size_t)r * ldc;
      *p = add ? (*p + acc[r]) : acc[r];
    }
  }
}

__device__ __forceinline__ float gelu_exact(float x) {
  return 0.5f * x * (1.0f + erff(x * 0.7071067811865476f));
}

// row[0:80] : row[d] += bias[d]; row = LN(row)*g + be   (eps = 1e-5)
__device__ __forceinline__ void bias_ln_row(float* row, const float* bias,
                                            const float* g, const float* be) {
  float m = 0.0f;
  for (int d = 0; d < DD; ++d) { row[d] += bias[d]; m += row[d]; }
  m *= (1.0f / DD);
  float var = 0.0f;
  for (int d = 0; d < DD; ++d) { const float t = row[d] - m; var += t * t; }
  var *= (1.0f / DD);
  const float inv = rsqrtf(var + 1e-5f);
  for (int d = 0; d < DD; ++d) row[d] = (row[d] - m) * inv * g[d] + be[d];
}

// ---------------------------------------------------------------------------
// Kernel 0: pack a weight matrix into the WMMA B-register layout (bf16).
// out[((l*ntiles+nt)*ksteps+ks)*512 + lane*16 + i] =
//     W[l][k][n] (or W[l][n][k] if transposed) * scale,  zero-padded.
//   k = ks*32 + (lane&16 ? 8 : 0) + (i < 8 ? i : i + 8),  n = nt*16 + (lane&15)
// ---------------------------------------------------------------------------
__global__ void k_pack(const float* __restrict__ W, int K, int N, int nlayers,
                       size_t lstride, int transposed, float scale,
                       __bf16* __restrict__ out) {
  const int ksteps = (K + 31) >> 5;
  const int ntiles = (N + 15) >> 4;
  const size_t total = (size_t)nlayers * ntiles * ksteps * 512;
  for (size_t gid = (size_t)blockIdx.x * blockDim.x + threadIdx.x; gid < total;
       gid += (size_t)gridDim.x * blockDim.x) {
    const int    i    = (int)(gid & 15);
    const int    lane = (int)((gid >> 4) & 31);
    const size_t ts   = gid >> 9;
    const int    ks   = (int)(ts % ksteps);
    const size_t t2   = ts / ksteps;
    const int    nt   = (int)(t2 % ntiles);
    const int    l    = (int)(t2 / ntiles);
    const int k = ks * 32 + ((lane & 16) ? 8 : 0) + (i < 8 ? i : i + 8);
    const int n = nt * 16 + (lane & 15);
    float v = 0.0f;
    if (k < K && n < N)
      v = transposed ? W[(size_t)l * lstride + (size_t)n * K + k]
                     : W[(size_t)l * lstride + (size_t)k * N + n];
    out[gid] = (__bf16)(v * scale);
  }
}

// ---------------------------------------------------------------------------
// Kernel 1: embeddings + input projection + LN + positional embedding
// ---------------------------------------------------------------------------
__global__ __launch_bounds__(160)
void k_embed(const int* __restrict__ loc, const int* __restrict__ usr,
             const int* __restrict__ wkd, const int* __restrict__ smin,
             const int* __restrict__ dur, const int* __restrict__ dif,
             const float* __restrict__ loc_t, const float* __restrict__ usr_t,
             const float* __restrict__ wk_t, const float* __restrict__ hr_t,
             const float* __restrict__ in_w, const float* __restrict__ in_b,
             const float* __restrict__ in_g, const float* __restrict__ in_be,
             const float* __restrict__ pos, float* __restrict__ xws) {
  const int tok = blockIdx.x;           // b*L + t
  const int b   = tok / LL;
  const int t   = tok - b * LL;
  const int tid = threadIdx.x;

  __shared__ float vec[INDIM];
  __shared__ float y[DD];

  if (tid < 80) {
    vec[tid] = loc_t[(size_t)loc[tok] * DD + tid];
  } else if (tid < 104) {
    vec[tid] = usr_t[(size_t)usr[(size_t)b * LL] * 24 + (tid - 80)];
  } else if (tid < 120) {
    vec[tid] = wk_t[(size_t)wkd[tok] * 16 + (tid - 104)];
  } else if (tid < 136) {
    int hr = smin[tok] / 60; hr = min(max(hr, 0), 23);
    vec[tid] = hr_t[(size_t)hr * 16 + (tid - 120)];
  } else if (tid == 136) {
    vec[tid] = (float)dif[tok] * 0.1f;
  } else if (tid == 137) {
    vec[tid] = (float)dur[tok] * (1.0f / 300.0f);
  }
  __syncthreads();

  if (tid < DD) {
    float acc = in_b[tid];
    for (int k = 0; k < INDIM; ++k) acc += vec[k] * in_w[(size_t)k * DD + tid];
    y[tid] = acc;
  }
  __syncthreads();

  if (tid < DD) {
    float m = 0.0f;
    for (int d = 0; d < DD; ++d) m += y[d];
    m *= (1.0f / DD);
    float var = 0.0f;
    for (int d = 0; d < DD; ++d) { const float u = y[d] - m; var += u * u; }
    var *= (1.0f / DD);
    const float inv = rsqrtf(var + 1e-5f);
    xws[(size_t)tok * DD + tid] =
        (y[tid] - m) * inv * in_g[tid] + in_be[tid] + pos[(size_t)t * DD + tid];
  }
}

// ---------------------------------------------------------------------------
// Kernel 2: normalize prototype rows -> pnorm[NLOC][80]
// ---------------------------------------------------------------------------
__global__ void k_protos(const float* __restrict__ protos,
                         float* __restrict__ pnorm) {
  const int r = blockIdx.x * blockDim.x + threadIdx.x;
  if (r >= NLOC) return;
  const float* src = protos + (size_t)r * DD;
  float ss = 0.0f;
  for (int d = 0; d < DD; ++d) ss += src[d] * src[d];
  const float inv = 1.0f / fmaxf(sqrtf(ss), 1e-12f);
  float* dst = pnorm + (size_t)r * DD;
  for (int d = 0; d < DD; ++d) dst[d] = src[d] * inv;
}

// ---------------------------------------------------------------------------
// Kernel 3: fused 3-layer transformer, one 256-thread block per sequence.
// Dynamic LDS (f32): xf[64][100] | qkvf[64][240] | of[64][100] | sc[64][64]
// ---------------------------------------------------------------------------
__global__ __launch_bounds__(256)
void k_layers(const float* __restrict__ xws, const int* __restrict__ loc,
              const int* __restrict__ slen,
              const __bf16* __restrict__ pk_qkv, const __bf16* __restrict__ pk_proj,
              const __bf16* __restrict__ pk_f1, const __bf16* __restrict__ pk_f2,
              const float* __restrict__ proj_b, const float* __restrict__ rel_emb,
              const float* __restrict__ n1_g, const float* __restrict__ n1_b,
              const float* __restrict__ ffn_b1, const float* __restrict__ ffn_b2,
              const float* __restrict__ n2_g, const float* __restrict__ n2_b,
              const float* __restrict__ fn_g, const float* __restrict__ fn_b,
              const float* __restrict__ pr_w1, const float* __restrict__ pr_b1,
              float* __restrict__ gws, float* __restrict__ fnws) {
  const int b    = blockIdx.x;
  const int tid  = threadIdx.x;
  const int wave = tid >> 5;
  const int lane = tid & 31;

  extern __shared__ float smem[];
  float* xf   = smem;                    // [64][AS]
  float* qkvf = xf + LP * AS;            // [64][240] (reused as ffn h1 [64][160])
  float* of   = qkvf + LP * 3 * DD;      // [64][AS]
  float* sc   = of + LP * AS;            // [64][64]
  __shared__ int   lm[LP];
  __shared__ float fin[DD];

  // load x, zero-padded (rows >= L and cols >= 80 stay exactly zero)
  for (int i = tid; i < LP * AS; i += 256) {
    const int r = i / AS, c = i - r * AS;
    xf[i] = (r < LL && c < DD) ? xws[((size_t)b * LL + r) * DD + c] : 0.0f;
  }
  if (tid < LP) lm[tid] = (tid < LL) ? loc[(size_t)b * LL + tid] : 0;
  __syncthreads();

  const float scale = 0.31622776601683794f;  // 1/sqrt(HD)

  for (int l = 0; l < NLAY; ++l) {
    // ---- qkv = x @ qkv_w[l]  (M=64, N=240, K=80->3 ksteps) ---------------
    gemm64p(xf, AS, (const v16bf*)pk_qkv + (size_t)l * 15 * 3 * 32, 15, 3,
            qkvf, 3 * DD, false, wave, lane);
    __syncthreads();

    // zero attention-out panel (keeps pad rows/cols exactly zero)
    for (int i = tid; i < LP * AS; i += 256) of[i] = 0.0f;

    // ---- attention (fp32 VALU; HD=10 too small for the matrix path) ------
    const float* rel = rel_emb + (size_t)l * (2 * LL - 1) * HDIM;
    for (int h = 0; h < HH; ++h) {
      for (int idx = tid; idx < LL * LP; idx += 256) {
        const int i = idx >> 6, j = idx & 63;
        float s = -INFINITY;
        if (j < LL && lm[j] != 0) {
          const float* qi = &qkvf[(size_t)i * 240 + h * HDIM];
          const float* kj = &qkvf[(size_t)j * 240 + DD + h * HDIM];
          const float* rj = rel + (size_t)(j - i + LL - 1) * HDIM;
          float acc = 0.0f;
#pragma unroll
          for (int d = 0; d < HDIM; ++d) acc += qi[d] * (kj[d] + rj[d]);
          s = acc * scale;
        }
        sc[idx] = s;
      }
      __syncthreads();
      if (tid < LL) {                      // softmax over row tid
        float* row = &sc[(size_t)tid * LP];
        float mx = -INFINITY;
        for (int j = 0; j < LL; ++j) mx = fmaxf(mx, row[j]);
        float sum = 0.0f;
        if (mx > -INFINITY)
          for (int j = 0; j < LL; ++j) { const float e = expf(row[j] - mx); row[j] = e; sum += e; }
        const float inv = (sum > 0.0f) ? 1.0f / sum : 0.0f;
        for (int j = 0; j < LL; ++j) row[j] = (mx > -INFINITY) ? row[j] * inv : 0.0f;
      }
      __syncthreads();
      for (int idx = tid; idx < LL * HDIM; idx += 256) {
        const int i = idx / HDIM, d = idx - i * HDIM;
        float acc = 0.0f;
        for (int j = 0; j < LL; ++j)
          acc += sc[(size_t)i * LP + j] * qkvf[(size_t)j * 240 + 2 * DD + h * HDIM + d];
        of[(size_t)i * AS + h * HDIM + d] = acc;
      }
      __syncthreads();
    }

    // ---- x += attn_out @ proj_w[l] ; LN1 ---------------------------------
    gemm64p(of, AS, (const v16bf*)pk_proj + (size_t)l * 5 * 3 * 32, 5, 3,
            xf, AS, true, wave, lane);
    __syncthreads();
    if (tid < LL)
      bias_ln_row(&xf[(size_t)tid * AS], proj_b + (size_t)l * DD,
                  n1_g + (size_t)l * DD, n1_b + (size_t)l * DD);
    __syncthreads();

    // ---- h1 = gelu(x @ ffn_w1 + b1)  (M=64, N=160, K=80) -----------------
    gemm64p(xf, AS, (const v16bf*)pk_f1 + (size_t)l * 10 * 3 * 32, 10, 3,
            qkvf, DFF, false, wave, lane);
    __syncthreads();
    for (int idx = tid; idx < LP * DFF; idx += 256) {
      const int i = idx / DFF;
      float v = 0.0f;
      if (i < LL) v = gelu_exact(qkvf[idx] + ffn_b1[(size_t)l * DFF + (idx - i * DFF)]);
      qkvf[idx] = v;                       // pad rows forced to zero
    }
    __syncthreads();

    // ---- x += h1 @ ffn_w2 ; LN2  (K=160 -> 5 exact ksteps) ---------------
    gemm64p(qkvf, DFF, (const v16bf*)pk_f2 + (size_t)l * 5 * 5 * 32, 5, 5,
            xf, AS, true, wave, lane);
    __syncthreads();
    if (tid < LL)
      bias_ln_row(&xf[(size_t)tid * AS], ffn_b2 + (size_t)l * DD,
                  n2_g + (size_t)l * DD, n2_b + (size_t)l * DD);
    __syncthreads();
  }

  // ---- final LN on selected token; precompute head inputs ----------------
  int fi = slen[b] - 1; fi = min(max(fi, 0), LL - 1);
  if (tid < DD) {
    const float* row = &xf[(size_t)fi * AS];
    float m = 0.0f;
    for (int d = 0; d < DD; ++d) m += row[d];
    m *= (1.0f / DD);
    float var = 0.0f;
    for (int d = 0; d < DD; ++d) { const float u = row[d] - m; var += u * u; }
    var *= (1.0f / DD);
    const float inv = rsqrtf(var + 1e-5f);
    fin[tid] = (row[tid] - m) * inv * fn_g[tid] + fn_b[tid];
  }
  __syncthreads();
  if (tid < DD) {
    float acc = pr_b1[tid];
    for (int k = 0; k < DD; ++k) acc += fin[k] * pr_w1[(size_t)k * DD + tid];
    gws[(size_t)b * DD + tid] = gelu_exact(acc);
    float ss = 0.0f;
    for (int k = 0; k < DD; ++k) ss += fin[k] * fin[k];
    const float nrm = fmaxf(sqrtf(ss), 1e-12f);
    fnws[(size_t)b * DD + tid] = fin[tid] / nrm;
  }
}

// ---------------------------------------------------------------------------
// Kernel 4: logits = (0.5*(g@pr_w2 + pr_b2) + 7.5*fnorm@pnorm^T) / clamp(temp)
// A panels staged zero-padded in LDS; B from packed bf16 (scales folded in).
// ---------------------------------------------------------------------------
__global__ __launch_bounds__(256)
void k_logits(const float* __restrict__ gws, const float* __restrict__ fnws,
              const __bf16* __restrict__ pk_w2, const __bf16* __restrict__ pk_pn,
              const float* __restrict__ pr_b2, const float* __restrict__ temp,
              float* __restrict__ out) {
  const int mt   = blockIdx.x;          // 0 .. B/16-1
  const int tid  = threadIdx.x;
  const int wave = tid >> 5;
  const int lane = tid & 31;
  const float inv_t = 1.0f / fminf(fmaxf(temp[0], 0.5f), 3.0f);
  const int NT = (NLOC + 15) >> 4;      // 75

  __shared__ float ag[16 * 96];
  __shared__ float af[16 * 96];
  for (int i = tid; i < 16 * 96; i += 256) {
    const int r = i / 96, c = i - r * 96;
    const size_t src = ((size_t)mt * 16 + r) * DD + c;
    ag[i] = (c < DD) ? gws[src] : 0.0f;
    af[i] = (c < DD) ? fnws[src] : 0.0f;
  }
  __syncthreads();

  for (int nt = wave; nt < NT; nt += 8) {
    v8f acc = {};
    const v16bf* b2 = (const v16bf*)pk_w2 + ((size_t)nt * 3) * 32 + lane;
    const v16bf* bp = (const v16bf*)pk_pn + ((size_t)nt * 3) * 32 + lane;
#pragma unroll
    for (int ks = 0; ks < 3; ++ks) {
      v16bf a = load_a(ag + ks * 32, 96, lane);
      acc = WMMA_BF16(a, b2[(size_t)ks * 32], acc);
    }
#pragma unroll
    for (int ks = 0; ks < 3; ++ks) {
      v16bf a = load_a(af + ks * 32, 96, lane);
      acc = WMMA_BF16(a, bp[(size_t)ks * 32], acc);
    }
    const int n  = nt * 16 + (lane & 15);
    const int mb = mt * 16 + ((lane & 16) ? 8 : 0);
    if (n < NLOC) {
      const float bias = 0.5f * pr_b2[n];
#pragma unroll
      for (int r = 0; r < 8; ++r)
        out[(size_t)(mb + r) * NLOC + n] = (acc[r] + bias) * inv_t;
    }
  }
}

// ---------------------------------------------------------------------------
// Launch
// ---------------------------------------------------------------------------
extern "C" void kernel_launch(void* const* d_in, const int* in_sizes, int n_in,
                              void* d_out, int out_size, void* d_ws, size_t ws_size,
                              hipStream_t stream) {
  const int*   loc     = (const int*)  d_in[0];
  const int*   usr     = (const int*)  d_in[1];
  const int*   wkd     = (const int*)  d_in[2];
  const int*   smin    = (const int*)  d_in[3];
  const int*   dur     = (const int*)  d_in[4];
  const int*   dif     = (const int*)  d_in[5];
  const int*   slen    = (const int*)  d_in[6];
  const float* loc_t   = (const float*)d_in[7];
  const float* usr_t   = (const float*)d_in[8];
  const float* wk_t    = (const float*)d_in[9];
  const float* hr_t    = (const float*)d_in[10];
  const float* in_w    = (const float*)d_in[11];
  const float* in_b    = (const float*)d_in[12];
  const float* in_g    = (const float*)d_in[13];
  const float* in_be   = (const float*)d_in[14];
  const float* pos     = (const float*)d_in[15];
  const float* qkv_w   = (const float*)d_in[16];
  const float* proj_w  = (const float*)d_in[17];
  const float* proj_b  = (const float*)d_in[18];
  const float* rel_emb = (const float*)d_in[19];
  const float* n1_g    = (const float*)d_in[20];
  const float* n1_b    = (const float*)d_in[21];
  const float* ffn_w1  = (const float*)d_in[22];
  const float* ffn_b1  = (const float*)d_in[23];
  const float* ffn_w2  = (const float*)d_in[24];
  const float* ffn_b2  = (const float*)d_in[25];
  const float* n2_g    = (const float*)d_in[26];
  const float* n2_b    = (const float*)d_in[27];
  const float* fn_g    = (const float*)d_in[28];
  const float* fn_b    = (const float*)d_in[29];
  const float* pr_w1   = (const float*)d_in[30];
  const float* pr_b1   = (const float*)d_in[31];
  const float* pr_w2   = (const float*)d_in[32];
  const float* pr_b2   = (const float*)d_in[33];
  const float* protos  = (const float*)d_in[34];
  const float* temp    = (const float*)d_in[35];
  float* out = (float*)d_out;

  // ---- workspace layout: f32 region then packed-bf16 region --------------
  float* xws  = (float*)d_ws;                          // [B][L][80]
  float* gws  = xws + (size_t)BB * LL * DD;            // [B][80]
  float* fnws = gws + (size_t)BB * DD;                 // [B][80]
  float* pnw  = fnws + (size_t)BB * DD;                // [NLOC][80]
  __bf16* pk_qkv  = (__bf16*)(pnw + (size_t)NLOC * DD);
  const size_t sz_qkv  = (size_t)NLAY * 15 * 3 * 512;  // 3 ksteps, 15 ntiles
  __bf16* pk_proj = pk_qkv + sz_qkv;
  const size_t sz_proj = (size_t)NLAY * 5 * 3 * 512;
  __bf16* pk_f1   = pk_proj + sz_proj;
  const size_t sz_f1   = (size_t)NLAY * 10 * 3 * 512;
  __bf16* pk_f2   = pk_f1 + sz_f1;
  const size_t sz_f2   = (size_t)NLAY * 5 * 5 * 512;
  __bf16* pk_w2   = pk_f2 + sz_f2;
  const size_t sz_head = (size_t)75 * 3 * 512;
  __bf16* pk_pn   = pk_w2 + sz_head;

  // ---- pack weights into WMMA B layout (bf16, scales folded in) ----------
  k_pack<<<(unsigned)((sz_qkv  + 255) / 256), 256, 0, stream>>>(
      qkv_w, DD, 3 * DD, NLAY, (size_t)DD * 3 * DD, 0, 1.0f, pk_qkv);
  k_pack<<<(unsigned)((sz_proj + 255) / 256), 256, 0, stream>>>(
      proj_w, DD, DD, NLAY, (size_t)DD * DD, 0, 1.0f, pk_proj);
  k_pack<<<(unsigned)((sz_f1   + 255) / 256), 256, 0, stream>>>(
      ffn_w1, DD, DFF, NLAY, (size_t)DD * DFF, 0, 1.0f, pk_f1);
  k_pack<<<(unsigned)((sz_f2   + 255) / 256), 256, 0, stream>>>(
      ffn_w2, DFF, DD, NLAY, (size_t)DFF * DD, 0, 1.0f, pk_f2);
  k_pack<<<(unsigned)((sz_head + 255) / 256), 256, 0, stream>>>(
      pr_w2, DD, NLOC, 1, 0, 0, 0.5f, pk_w2);

  k_protos<<<(NLOC + 255) / 256, 256, 0, stream>>>(protos, pnw);
  k_pack<<<(unsigned)((sz_head + 255) / 256), 256, 0, stream>>>(
      pnw, DD, NLOC, 1, 0, /*transposed=*/1, 7.5f, pk_pn);

  k_embed<<<BB * LL, 160, 0, stream>>>(loc, usr, wkd, smin, dur, dif,
                                       loc_t, usr_t, wk_t, hr_t,
                                       in_w, in_b, in_g, in_be, pos, xws);

  const size_t lds_bytes =
      (size_t)(LP * AS + LP * 3 * DD + LP * AS + LP * LP) * sizeof(float);
  k_layers<<<BB, 256, lds_bytes, stream>>>(
      xws, loc, slen, pk_qkv, pk_proj, pk_f1, pk_f2,
      proj_b, rel_emb, n1_g, n1_b, ffn_b1, ffn_b2, n2_g, n2_b,
      fn_g, fn_b, pr_w1, pr_b1, gws, fnws);

  k_logits<<<BB / 16, 256, 0, stream>>>(gws, fnws, pk_w2, pk_pn, pr_b2, temp, out);
}
